// HyperSETBlock_71614284694153
// MI455X (gfx1250) — compile-verified, matrix operand order checked by reading, not run on hardware
//
#include <hip/hip_runtime.h>
#include <math.h>

// ---------------- problem constants ----------------
#define BB    16
#define TT    256
#define DIMD  768
#define HEADS 12
#define DHD   64
#define HID   3072
#define TDIM  1536
#define NREC  12
#define ROWS  (BB*TT)      // 4096 tokens
#define NBH   (BB*HEADS)   // 192 (b,h) pairs
#define EPSF   1.1920929e-07f
#define SCALEF 0.125f      // DH^-0.5

typedef __bf16 bf16;
typedef __attribute__((ext_vector_type(16))) __bf16 v16bf;
typedef __attribute__((ext_vector_type(8)))  __bf16 v8bf;
typedef __attribute__((ext_vector_type(8)))  float  v8f;

#if defined(__AMDGCN__) && __has_builtin(__builtin_amdgcn_global_load_async_to_lds_b128) && __has_builtin(__builtin_amdgcn_s_wait_asynccnt)
#define USE_ASYNC_LDS 1
// builtin expects: (v4i global*, lds ptr, imm offset, imm cpol); param pointee is a
// GCC-style vector_size(16) int (clang prints it as 'vector_size(4*sizeof(int)) int __device__*')
typedef int v4i __attribute__((vector_size(16)));
typedef __attribute__((address_space(1))) v4i gv4i;   // global (__device__) int4
typedef __attribute__((address_space(3))) v4i lv4i;   // LDS int4
#else
#define USE_ASYNC_LDS 0
#endif

__device__ __forceinline__ bf16 f2bf(float f) {
  unsigned u = __builtin_bit_cast(unsigned, f);
  u += 0x7FFFu + ((u >> 16) & 1u);          // round-to-nearest-even
  return __builtin_bit_cast(bf16, (unsigned short)(u >> 16));
}

// ---------------- generic batched bf16 WMMA GEMM ----------------
// C[b][m][n] = scale * sum_k A[b][m][k] * Bt[b][n][k]  (+ bias[n])
// A row-major [M][K] bf16, Bt row-major [N][K] bf16 (B transposed), C f32.
// Requires: M,K multiples of tile/K-step; ragged N handled by clamped loads
// (garbage columns never stored).
#define TM 128
#define TN 128
#define TKk 32
#define LDT 48   // row stride in halves: 96B, 16B-aligned sublanes

__global__ __launch_bounds__(256) void k_gemm(
    const bf16* __restrict__ A, const bf16* __restrict__ Bt,
    float* __restrict__ C,
    int M, int N, int K,
    long aBatch, long bBatch, long cBatch,
    float scale, const float* __restrict__ bias)
{
  __shared__ bf16 As[2][TM][LDT];
  __shared__ bf16 Bs[2][TN][LDT];

  const int bz = blockIdx.z;
  A  += (long)bz * aBatch;
  Bt += (long)bz * bBatch;
  C  += (long)bz * cBatch;

  const int m0 = blockIdx.y * TM;
  const int n0 = blockIdx.x * TN;
  const int tid  = threadIdx.x;
  const int lane = tid & 31;
  const int wave = tid >> 5;
  const int wm = wave >> 1;   // 0..3 -> 32-row band
  const int wn = wave & 1;    // 0..1 -> 64-col band

  // ---- loop-invariant staging coordinates (clamped: no exec-mask branches) ----
  const int srow = tid >> 2;             // 0..63
  const int cc   = (tid & 3) << 3;       // 0,8,16,24 halves
  int ar0 = m0 + srow;      if (ar0 > M - 1) ar0 = M - 1;
  int ar1 = m0 + srow + 64; if (ar1 > M - 1) ar1 = M - 1;
  int br0 = n0 + srow;      if (br0 > N - 1) br0 = N - 1;
  int br1 = n0 + srow + 64; if (br1 > N - 1) br1 = N - 1;
  const bf16* aP0 = A  + (size_t)ar0 * K + cc;
  const bf16* aP1 = A  + (size_t)ar1 * K + cc;
  const bf16* bP0 = Bt + (size_t)br0 * K + cc;
  const bf16* bP1 = Bt + (size_t)br1 * K + cc;

  v8f acc[2][4];
  #pragma unroll
  for (int i = 0; i < 2; ++i)
    #pragma unroll
    for (int j = 0; j < 4; ++j)
      #pragma unroll
      for (int r = 0; r < 8; ++r) acc[i][j][r] = 0.0f;

  const int nk = K / TKk;

  // ---- prologue: stage tile 0 into buffer 0 ----
#if USE_ASYNC_LDS
  __builtin_amdgcn_global_load_async_to_lds_b128((gv4i*)aP0, (lv4i*)&As[0][srow][cc], 0, 0);
  __builtin_amdgcn_global_load_async_to_lds_b128((gv4i*)aP1, (lv4i*)&As[0][srow + 64][cc], 0, 0);
  __builtin_amdgcn_global_load_async_to_lds_b128((gv4i*)bP0, (lv4i*)&Bs[0][srow][cc], 0, 0);
  __builtin_amdgcn_global_load_async_to_lds_b128((gv4i*)bP1, (lv4i*)&Bs[0][srow + 64][cc], 0, 0);
#else
  {
    v8bf ra0 = *reinterpret_cast<const v8bf*>(aP0);
    v8bf ra1 = *reinterpret_cast<const v8bf*>(aP1);
    v8bf rb0 = *reinterpret_cast<const v8bf*>(bP0);
    v8bf rb1 = *reinterpret_cast<const v8bf*>(bP1);
    *reinterpret_cast<v8bf*>(&As[0][srow][cc])      = ra0;
    *reinterpret_cast<v8bf*>(&As[0][srow + 64][cc]) = ra1;
    *reinterpret_cast<v8bf*>(&Bs[0][srow][cc])      = rb0;
    *reinterpret_cast<v8bf*>(&Bs[0][srow + 64][cc]) = rb1;
  }
#endif

  const int fr   = lane & 15;
  const int koff = (lane >> 4) << 3;   // 0 or 8

  for (int kt = 0; kt < nk; ++kt) {
    const int cur = kt & 1;
    const int nxt = cur ^ 1;
    const int kn  = (kt + 1) * TKk;

#if USE_ASYNC_LDS
    __builtin_amdgcn_s_wait_asynccnt(0);   // tile kt landed in LDS
    __syncthreads();
    if (kt + 1 < nk) {                     // overlap: fetch tile kt+1 during compute
      __builtin_amdgcn_global_load_async_to_lds_b128((gv4i*)(aP0 + kn), (lv4i*)&As[nxt][srow][cc], 0, 0);
      __builtin_amdgcn_global_load_async_to_lds_b128((gv4i*)(aP1 + kn), (lv4i*)&As[nxt][srow + 64][cc], 0, 0);
      __builtin_amdgcn_global_load_async_to_lds_b128((gv4i*)(bP0 + kn), (lv4i*)&Bs[nxt][srow][cc], 0, 0);
      __builtin_amdgcn_global_load_async_to_lds_b128((gv4i*)(bP1 + kn), (lv4i*)&Bs[nxt][srow + 64][cc], 0, 0);
    }
#else
    __syncthreads();
    v8bf ra0, ra1, rb0, rb1;
    if (kt + 1 < nk) {                     // prefetch next tile into registers
      ra0 = *reinterpret_cast<const v8bf*>(aP0 + kn);
      ra1 = *reinterpret_cast<const v8bf*>(aP1 + kn);
      rb0 = *reinterpret_cast<const v8bf*>(bP0 + kn);
      rb1 = *reinterpret_cast<const v8bf*>(bP1 + kn);
    }
#endif

    // ---- fragment loads per CDNA5 16-bit layout + 8 WMMAs ----
    v16bf af[2], bfv[4];
    #pragma unroll
    for (int i = 0; i < 2; ++i) {
      const bf16* p = &As[cur][wm * 32 + i * 16 + fr][koff];
      v8bf lo = *reinterpret_cast<const v8bf*>(p);
      v8bf hi = *reinterpret_cast<const v8bf*>(p + 16);
      af[i] = __builtin_shufflevector(lo, hi, 0,1,2,3,4,5,6,7,8,9,10,11,12,13,14,15);
    }
    #pragma unroll
    for (int j = 0; j < 4; ++j) {
      const bf16* p = &Bs[cur][wn * 64 + j * 16 + fr][koff];
      v8bf lo = *reinterpret_cast<const v8bf*>(p);
      v8bf hi = *reinterpret_cast<const v8bf*>(p + 16);
      bfv[j] = __builtin_shufflevector(lo, hi, 0,1,2,3,4,5,6,7,8,9,10,11,12,13,14,15);
    }
    #pragma unroll
    for (int i = 0; i < 2; ++i)
      #pragma unroll
      for (int j = 0; j < 4; ++j)
        acc[i][j] = __builtin_amdgcn_wmma_f32_16x16x32_bf16(
            false, af[i], false, bfv[j], (short)0, acc[i][j], false, false);

#if !USE_ASYNC_LDS
    if (kt + 1 < nk) {                     // drain registers into the other buffer
      *reinterpret_cast<v8bf*>(&As[nxt][srow][cc])      = ra0;
      *reinterpret_cast<v8bf*>(&As[nxt][srow + 64][cc]) = ra1;
      *reinterpret_cast<v8bf*>(&Bs[nxt][srow][cc])      = rb0;
      *reinterpret_cast<v8bf*>(&Bs[nxt][srow + 64][cc]) = rb1;
    }
#endif
  }

  // ---- epilogue: f32 store per C/D layout (lane<16: M=r, lane>=16: M=r+8) ----
  const int nl    = lane & 15;
  const int rbase = (lane >> 4) << 3;
  #pragma unroll
  for (int i = 0; i < 2; ++i) {
    #pragma unroll
    for (int j = 0; j < 4; ++j) {
      int gn = n0 + wn * 64 + j * 16 + nl;
      if (gn < N) {
        float bv = bias ? bias[gn] : 0.0f;
        #pragma unroll
        for (int r = 0; r < 8; ++r) {
          int gm = m0 + wm * 32 + i * 16 + rbase + r;
          if (gm < M) C[(size_t)gm * N + gn] = acc[i][j][r] * scale + bv;
        }
      }
    }
  }
}

// ---------------- elementwise / reduction kernels ----------------

__global__ void k_cvt(const float* __restrict__ s, bf16* __restrict__ d, int n) {
  int gid = blockIdx.x * 256 + threadIdx.x;
  if (gid < n) d[gid] = f2bf(s[gid]);
}

__global__ void k_transpose(const float* __restrict__ s, bf16* __restrict__ d, int R, int C) {
  int gid = blockIdx.x * 256 + threadIdx.x;
  if (gid < R * C) {
    int r = gid / C, c = gid % C;
    d[(size_t)c * R + r] = f2bf(s[gid]);
  }
}

// g = gelu(rmsnorm(recur_emb(step) + x_inp, ln_t_w))   -> bf16 [4096][768]
__global__ __launch_bounds__(256) void k_time_pre(const float* __restrict__ x,
                                                  const float* __restrict__ lntw,
                                                  bf16* __restrict__ g, float tval) {
  __shared__ float red[256];
  const int row = blockIdx.x;
  const int tid = threadIdx.x;
  const float kfreq = -0.024047886f;  // -ln(10000)/383
  float v[3]; float ss = 0.0f;
  #pragma unroll
  for (int i = 0; i < 3; ++i) {
    int d = tid + i * 256;
    float c;
    if (d < 384) c = __sinf(tval * __expf(kfreq * (float)d));
    else         c = __cosf(tval * __expf(kfreq * (float)(d - 384)));
    v[i] = x[(size_t)row * DIMD + d] + c;
    ss += v[i] * v[i];
  }
  red[tid] = ss; __syncthreads();
  for (int s = 128; s > 0; s >>= 1) { if (tid < s) red[tid] += red[tid + s]; __syncthreads(); }
  float inv = rsqrtf(red[0] / (float)DIMD + EPSF);
  #pragma unroll
  for (int i = 0; i < 3; ++i) {
    int d = tid + i * 256;
    float t = v[i] * inv * lntw[d];
    float ge = 0.5f * t * (1.0f + erff(t * 0.70710678f));
    g[(size_t)row * DIMD + d] = f2bf(ge);
  }
}

// per-(token,head) rmsnorm over 64; emit w_bf[bh][t][d] and wT_bf[bh][d][t]
__global__ __launch_bounds__(64) void k_head_rmsnorm(const float* __restrict__ w,
                                                     const float* __restrict__ ln1w,
                                                     bf16* __restrict__ wbf,
                                                     bf16* __restrict__ wtbf) {
  __shared__ float red[64];
  const int ht = blockIdx.x;
  const int token = ht / HEADS, h = ht % HEADS;
  const int b = token >> 8, t = token & 255;
  const int d = threadIdx.x;
  float val = w[(size_t)token * DIMD + h * DHD + d];
  red[d] = val * val; __syncthreads();
  for (int s = 32; s > 0; s >>= 1) { if (d < s) red[d] += red[d + s]; __syncthreads(); }
  float inv = rsqrtf(red[0] / (float)DHD + EPSF);
  bf16 o = f2bf(val * inv * ln1w[d]);
  int bh = b * HEADS + h;
  wbf[((size_t)bh * TT + t) * DHD + d]  = o;
  wtbf[((size_t)bh * DHD + d) * TT + t] = o;
}

// row/col max & exp-sums of S[bh][256][256]
__global__ __launch_bounds__(256) void k_attn_stats(const float* __restrict__ S,
                                                    float* __restrict__ st) {
  const int bh = blockIdx.x, t = threadIdx.x;
  const float* Sb = S + (size_t)bh * TT * TT;
  float rm = -3.0e38f, cm = -3.0e38f;
  for (int j = 0; j < TT; ++j) {
    rm = fmaxf(rm, Sb[(size_t)t * TT + j]);
    cm = fmaxf(cm, Sb[(size_t)j * TT + t]);
  }
  float rs = 0.0f, cs = 0.0f;
  for (int j = 0; j < TT; ++j) {
    rs += __expf(Sb[(size_t)t * TT + j] - rm);
    cs += __expf(Sb[(size_t)j * TT + t] - cm);
  }
  float* sb = st + (size_t)bh * 4 * TT;
  sb[t] = rm; sb[TT + t] = rs; sb[2 * TT + t] = cm; sb[3 * TT + t] = cs;
}

// A = softmax_row + softmax_col, bf16
__global__ __launch_bounds__(256) void k_form_A(const float* __restrict__ S,
                                                const float* __restrict__ st,
                                                bf16* __restrict__ Abf) {
  const int bh = blockIdx.x >> 8;
  const int i  = blockIdx.x & 255;
  const int j  = threadIdx.x;
  const float* sb = st + (size_t)bh * 4 * TT;
  size_t idx = ((size_t)bh * TT + i) * TT + j;
  float s = S[idx];
  float a = __expf(s - sb[i]) / sb[TT + i] + __expf(s - sb[2 * TT + j]) / sb[3 * TT + j];
  Abf[idx] = f2bf(a);
}

// O[bh][t][d] f32 -> o_bf[(b*T+t)*DIM + h*64+d]
__global__ void k_merge_heads(const float* __restrict__ O, bf16* __restrict__ obf) {
  int gid = blockIdx.x * 256 + threadIdx.x;
  if (gid >= ROWS * DIMD) return;
  int row = gid / DIMD, col = gid % DIMD;
  int b = row >> 8, t = row & 255, h = col >> 6, d = col & 63;
  obf[gid] = f2bf(O[(((size_t)(b * HEADS + h)) * TT + t) * DHD + d]);
}

// x1 = xc - alpha * proj ; alpha = te[:, :768]
__global__ void k_combine1(const float* __restrict__ xc, const float* __restrict__ te,
                           const float* __restrict__ proj,
                           float* __restrict__ x1, bf16* __restrict__ x1bf) {
  int gid = blockIdx.x * 256 + threadIdx.x;
  if (gid >= ROWS * DIMD) return;
  int row = gid / DIMD, col = gid % DIMD;
  float a = te[(size_t)row * TDIM + col];
  float v = xc[gid] - a * proj[gid];
  x1[gid] = v; x1bf[gid] = f2bf(v);
}

// x = x1 + beta * down ; beta = te[:, 768:]; also emit bf16 for next step's qkv
__global__ void k_combine2(const float* __restrict__ x1, const float* __restrict__ te,
                           const float* __restrict__ dn,
                           float* __restrict__ dst, bf16* __restrict__ xcbf) {
  int gid = blockIdx.x * 256 + threadIdx.x;
  if (gid >= ROWS * DIMD) return;
  int row = gid / DIMD, col = gid % DIMD;
  float bta = te[(size_t)row * TDIM + DIMD + col];
  float v = x1[gid] + bta * dn[gid];
  dst[gid] = v; xcbf[gid] = f2bf(v);
}

// h_bf = relu(rmsnorm(h, ln2_w)) over 3072 per token
__global__ __launch_bounds__(256) void k_rmsrelu(const float* __restrict__ h,
                                                 const float* __restrict__ ln2w,
                                                 bf16* __restrict__ hbf) {
  __shared__ float red[256];
  const int row = blockIdx.x;
  const int tid = threadIdx.x;
  float v[12]; float ss = 0.0f;
  #pragma unroll
  for (int i = 0; i < 12; ++i) {
    int d = tid + i * 256;
    v[i] = h[(size_t)row * HID + d];
    ss += v[i] * v[i];
  }
  red[tid] = ss; __syncthreads();
  for (int s = 128; s > 0; s >>= 1) { if (tid < s) red[tid] += red[tid + s]; __syncthreads(); }
  float inv = rsqrtf(red[0] / (float)HID + EPSF);
  #pragma unroll
  for (int i = 0; i < 12; ++i) {
    int d = tid + i * 256;
    float t = v[i] * inv * ln2w[d];
    hbf[(size_t)row * HID + d] = f2bf(fmaxf(t, 0.0f));
  }
}

// ---------------- host driver ----------------
extern "C" void kernel_launch(void* const* d_in, const int* in_sizes, int n_in,
                              void* d_out, int out_size, void* d_ws, size_t ws_size,
                              hipStream_t stream) {
  (void)in_sizes; (void)n_in; (void)out_size; (void)ws_size;
  const float* x_in = (const float*)d_in[0];
  const float* Wqkv = (const float*)d_in[1];
  const float* ln1w = (const float*)d_in[2];
  const float* Wmlp = (const float*)d_in[3];
  const float* ln2w = (const float*)d_in[4];
  const float* lntw = (const float*)d_in[5];
  const float* Wt   = (const float*)d_in[6];
  const float* bt   = (const float*)d_in[7];
  float* out = (float*)d_out;

  char* ws = (char*)d_ws;
  size_t off = 0;
  auto alloc = [&](size_t bytes) -> char* {
    char* p = ws + off; off += (bytes + 255) & ~(size_t)255; return p;
  };

  bf16*  w_qkv_bf  = (bf16*) alloc((size_t)DIMD*DIMD*2);
  bf16*  w_qkvT_bf = (bf16*) alloc((size_t)DIMD*DIMD*2);
  bf16*  w_mlp_bf  = (bf16*) alloc((size_t)HID*DIMD*2);
  bf16*  w_mlpT_bf = (bf16*) alloc((size_t)HID*DIMD*2);
  bf16*  w_t_bf    = (bf16*) alloc((size_t)TDIM*DIMD*2);
  bf16*  xc_bf     = (bf16*) alloc((size_t)ROWS*DIMD*2);
  bf16*  g_bf      = (bf16*) alloc((size_t)ROWS*DIMD*2);
  float* te        = (float*)alloc((size_t)ROWS*TDIM*4);
  float* w_f32     = (float*)alloc((size_t)ROWS*DIMD*4);     // qkv out, reused for proj out
  bf16*  w_bf      = (bf16*) alloc((size_t)NBH*TT*DHD*2);
  bf16*  wT_bf     = (bf16*) alloc((size_t)NBH*DHD*TT*2);
  float* Sb        = (float*)alloc((size_t)NBH*TT*TT*4);     // == 4096*3072*4 -> reused as h_f32
  float* stats     = (float*)alloc((size_t)NBH*4*TT*4);
  bf16*  A_bf      = (bf16*) alloc((size_t)NBH*TT*TT*2);     // == 4096*3072*2 -> reused as h_bf
  float* O_f32     = (float*)alloc((size_t)NBH*TT*DHD*4);    // == 4096*768*4 -> reused as down
  bf16*  o_bf      = (bf16*) alloc((size_t)ROWS*DIMD*2);
  float* x1_f32    = (float*)alloc((size_t)ROWS*DIMD*4);
  bf16*  x1_bf     = (bf16*) alloc((size_t)ROWS*DIMD*2);
  float* x_cur     = (float*)alloc((size_t)ROWS*DIMD*4);

  float* h_f32   = Sb;            // buffer reuse across phases
  bf16*  h_bf    = A_bf;
  float* down    = O_f32;

  const int EW = ROWS * DIMD;     // 3,145,728

  // weight conversions (deterministic, every call)
  k_cvt      <<<(DIMD*DIMD+255)/256, 256, 0, stream>>>(Wqkv, w_qkv_bf, DIMD*DIMD);
  k_transpose<<<(DIMD*DIMD+255)/256, 256, 0, stream>>>(Wqkv, w_qkvT_bf, DIMD, DIMD);
  k_cvt      <<<(HID*DIMD+255)/256, 256, 0, stream>>>(Wmlp, w_mlp_bf, HID*DIMD);
  k_transpose<<<(HID*DIMD+255)/256, 256, 0, stream>>>(Wmlp, w_mlpT_bf, HID, DIMD);
  k_cvt      <<<(TDIM*DIMD+255)/256, 256, 0, stream>>>(Wt, w_t_bf, TDIM*DIMD);
  k_cvt      <<<(EW+255)/256, 256, 0, stream>>>(x_in, xc_bf, EW);

  for (int step = 0; step < NREC; ++step) {
    float tval = (float)(step + 1);
    const float* xc_f32 = (step == 0) ? x_in : x_cur;

    // time MLP: te = gelu(rmsnorm(c + x_inp)) @ Wt^T + bt
    k_time_pre<<<ROWS, 256, 0, stream>>>(x_in, lntw, g_bf, tval);
    k_gemm<<<dim3(TDIM/128, ROWS/128, 1), 256, 0, stream>>>(
        g_bf, w_t_bf, te, ROWS, TDIM, DIMD, 0, 0, 0, 1.0f, bt);

    // qkv + per-head rmsnorm
    k_gemm<<<dim3(DIMD/128, ROWS/128, 1), 256, 0, stream>>>(
        xc_bf, w_qkv_bf, w_f32, ROWS, DIMD, DIMD, 0, 0, 0, 1.0f, nullptr);
    k_head_rmsnorm<<<ROWS*HEADS, 64, 0, stream>>>(w_f32, ln1w, w_bf, wT_bf);

    // scores, bisoftmax, O = A @ w
    k_gemm<<<dim3(2, 2, NBH), 256, 0, stream>>>(
        w_bf, w_bf, Sb, TT, TT, DHD,
        (long)TT*DHD, (long)TT*DHD, (long)TT*TT, SCALEF, nullptr);
    k_attn_stats<<<NBH, 256, 0, stream>>>(Sb, stats);
    k_form_A<<<NBH*TT, 256, 0, stream>>>(Sb, stats, A_bf);
    k_gemm<<<dim3(1, 2, NBH), 256, 0, stream>>>(
        A_bf, wT_bf, O_f32, TT, DHD, TT,
        (long)TT*TT, (long)DHD*TT, (long)TT*DHD, 1.0f, nullptr);
    k_merge_heads<<<(EW+255)/256, 256, 0, stream>>>(O_f32, o_bf);

    // projection + x1 = xc - alpha * attn
    k_gemm<<<dim3(DIMD/128, ROWS/128, 1), 256, 0, stream>>>(
        o_bf, w_qkvT_bf, w_f32, ROWS, DIMD, DIMD, 0, 0, 0, 1.0f, nullptr);
    k_combine1<<<(EW+255)/256, 256, 0, stream>>>(xc_f32, te, w_f32, x1_f32, x1_bf);

    // MLP: up -> rmsnorm+relu -> down ; x = x1 + beta * ff
    k_gemm<<<dim3(HID/128, ROWS/128, 1), 256, 0, stream>>>(
        x1_bf, w_mlp_bf, h_f32, ROWS, HID, DIMD, 0, 0, 0, 1.0f, nullptr);
    k_rmsrelu<<<ROWS, 256, 0, stream>>>(h_f32, ln2w, h_bf);
    k_gemm<<<dim3(DIMD/128, ROWS/128, 1), 256, 0, stream>>>(
        h_bf, w_mlpT_bf, down, ROWS, DIMD, HID, 0, 0, 0, 1.0f, nullptr);

    float* dst = (step == NREC - 1) ? out : x_cur;
    k_combine2<<<(EW+255)/256, 256, 0, stream>>>(x1_f32, te, down, dst, xc_bf);
  }
}
